// HiddenLayer_7730941133085
// MI455X (gfx1250) — matrix operand order repaired and checked by reading, not compiled, not run on hardware
//
#include <hip/hip_runtime.h>
#include <hip/hip_bf16.h>
#include <stdint.h>

typedef __attribute__((ext_vector_type(16))) _Float16 v16h;
typedef __attribute__((ext_vector_type(8)))  _Float16 v8h;
typedef __attribute__((ext_vector_type(4)))  _Float16 v4h;
typedef __attribute__((ext_vector_type(8)))  float    v8f;
typedef __attribute__((ext_vector_type(4)))  unsigned int u32x4;
typedef __attribute__((ext_vector_type(8)))  int          i32x8;
typedef __attribute__((ext_vector_type(4)))  int          i32x4;

#define BDIM   8192
#define INDIM  1024
#define OUTDIM 4096

#define BM 128
#define BN 64
#define BK 64
#define LDP 72   // padded LDS row stride in halves (144 B -> rows hit distinct bank groups)
#define NKSTEP (INDIM / BK)

#if defined(__has_builtin)
#  if __has_builtin(__builtin_amdgcn_tensor_load_to_lds) && __has_builtin(__builtin_amdgcn_s_wait_tensorcnt)
#    define USE_TDM 1
#  else
#    define USE_TDM 0
#  endif
#else
#  define USE_TDM 0
#endif

union frag16 { v16h v; v8h h[2]; };

__device__ __forceinline__ float f_nmos(float v) {
    float t = fmaxf(v - 0.2f, 0.0f);
    return 2.5f * t * t;
}

__device__ __forceinline__ float g_pmos(float i) {
    float g = 1.2f * (1.0f - __expf(-0.05f * i));
    return fminf(fmaxf(g, 0.0f), 1.2f);
}

#if USE_TDM
// Issue one TDM 2-D tile load: rows x BK f16 elements, row stride INDIM elements,
// into LDS with a 16-byte pad after every 128-byte row (=> LDP halves stride).
__device__ __forceinline__ void tdm_load_tile(const _Float16* gsrc, unsigned lds_byte_off,
                                              unsigned tensor_rows, unsigned tile_rows) {
    unsigned long long ga = (unsigned long long)(size_t)gsrc;
    u32x4 g0;
    g0[0] = 1u;                                                   // count=1, user descriptor
    g0[1] = lds_byte_off;                                         // lds_addr (bytes)
    g0[2] = (unsigned)ga;                                         // global_addr[31:0]
    g0[3] = (unsigned)((ga >> 32) & 0x01FFFFFFull) | 0x80000000u; // global_addr[56:32] | type=2
    i32x8 g1;
    g1[0] = (int)((1u << 16)    // data_size = 2 bytes (f16)
                | (1u << 20)    // pad_enable
                | (4u << 22)    // pad_interval: every 32 DWORDs (= one 128 B tile row)
                | (3u << 25));  // pad_amount : 4 DWORDs (= 16 B)
    g1[1] = (int)((unsigned)INDIM << 16);     // tensor_dim0[15:0] in [31:16]
    g1[2] = (int)(tensor_rows << 16);         // tensor_dim0 hi=0 | tensor_dim1[15:0]
    g1[3] = (int)((unsigned)BK << 16);        // tensor_dim1 hi=0 | tile_dim0 = 64
    g1[4] = (int)tile_rows;                   // tile_dim1 | tile_dim2 = 0
    g1[5] = (int)INDIM;                       // tensor_dim0_stride low 32 (elements)
    g1[6] = 0;
    g1[7] = 0;
    i32x4 z4; z4[0] = z4[1] = z4[2] = z4[3] = 0;
    i32x8 z8; z8[0] = z8[1] = z8[2] = z8[3] = z8[4] = z8[5] = z8[6] = z8[7] = 0;
    // 6-arg form (clang-23 / therock-10.0 headers)
    __builtin_amdgcn_tensor_load_to_lds(g0, g1, z4, z4, z8, 0);
}
#define TWAIT(n) __builtin_amdgcn_s_wait_tensorcnt(n)
#endif

// ---------------------------------------------------------------------------
// Precompute kernels: x -> f16 currents ; w -> f16 relu(+wq)/relu(-wq) pair
// ---------------------------------------------------------------------------
__global__ __launch_bounds__(256) void prep_x_kernel(const float* __restrict__ x,
                                                     _Float16* __restrict__ ih) {
    int idx = blockIdx.x * 256 + threadIdx.x;
    float4 v = ((const float4*)x)[idx];
    v4h h;
    h[0] = (_Float16)f_nmos(v.x);
    h[1] = (_Float16)f_nmos(v.y);
    h[2] = (_Float16)f_nmos(v.z);
    h[3] = (_Float16)f_nmos(v.w);
    ((v4h*)ih)[idx] = h;
}

__global__ __launch_bounds__(256) void prep_w_kernel(const float* __restrict__ w,
                                                     _Float16* __restrict__ wp,
                                                     _Float16* __restrict__ wn) {
    int idx = blockIdx.x * 256 + threadIdx.x;
    float4 v = ((const float4*)w)[idx];
    v4h hp, hn;
    float q;
    q = fminf(fmaxf(rintf(v.x), -8.0f), 7.0f);
    hp[0] = (_Float16)fmaxf(q, 0.0f); hn[0] = (_Float16)fmaxf(-q, 0.0f);
    q = fminf(fmaxf(rintf(v.y), -8.0f), 7.0f);
    hp[1] = (_Float16)fmaxf(q, 0.0f); hn[1] = (_Float16)fmaxf(-q, 0.0f);
    q = fminf(fmaxf(rintf(v.z), -8.0f), 7.0f);
    hp[2] = (_Float16)fmaxf(q, 0.0f); hn[2] = (_Float16)fmaxf(-q, 0.0f);
    q = fminf(fmaxf(rintf(v.w), -8.0f), 7.0f);
    hp[3] = (_Float16)fmaxf(q, 0.0f); hn[3] = (_Float16)fmaxf(-q, 0.0f);
    ((v4h*)wp)[idx] = hp;
    ((v4h*)wn)[idx] = hn;
}

// ---------------------------------------------------------------------------
// Main WMMA GEMM from precomputed f16 operands, TDM double-buffered staging
// ---------------------------------------------------------------------------
__global__ __launch_bounds__(256) void nmos_wmma_gemm_f16(const _Float16* __restrict__ ih,
                                                          const _Float16* __restrict__ wp,
                                                          const _Float16* __restrict__ wn,
                                                          float* __restrict__ out) {
    __shared__ __align__(16) _Float16 As [2][BM * LDP];
    __shared__ __align__(16) _Float16 Bps[2][BN * LDP];
    __shared__ __align__(16) _Float16 Bns[2][BN * LDP];

    const int t    = threadIdx.x;
    const int lane = t & 31;
    const int wv   = t >> 5;      // wave 0..7
    const int wmi  = wv >> 1;     // 0..3 : 32-row strip
    const int wni  = wv & 1;      // 0..1 : 32-col strip
    const int half = lane >> 4;
    const int lrow = lane & 15;

    const int row0 = blockIdx.y * BM;
    const int col0 = blockIdx.x * BN;

    const _Float16* aT = ih + (size_t)row0 * INDIM;
    const _Float16* pT = wp + (size_t)col0 * INDIM;
    const _Float16* nT = wn + (size_t)col0 * INDIM;

    v8f accP[2][2], accN[2][2];
#pragma unroll
    for (int mt = 0; mt < 2; ++mt)
#pragma unroll
        for (int nt = 0; nt < 2; ++nt)
#pragma unroll
            for (int r = 0; r < 8; ++r) {
                accP[mt][nt][r] = 0.0f;
                accN[mt][nt][r] = 0.0f;
            }

    auto stage = [&](int buf, int k0) {
#if USE_TDM
        if (wv == 0) {
            tdm_load_tile(aT + k0, (unsigned)(size_t)&As[buf][0],  BDIM,   BM);
            tdm_load_tile(pT + k0, (unsigned)(size_t)&Bps[buf][0], OUTDIM, BN);
            tdm_load_tile(nT + k0, (unsigned)(size_t)&Bns[buf][0], OUTDIM, BN);
        }
#else
#pragma unroll
        for (int it = 0; it < 4; ++it) {
            int v = it * 256 + t, r = v >> 3, c = (v & 7) << 3;
            *(v8h*)&As[buf][r * LDP + c] = *(const v8h*)(aT + (size_t)r * INDIM + k0 + c);
        }
#pragma unroll
        for (int it = 0; it < 2; ++it) {
            int v = it * 256 + t, r = v >> 3, c = (v & 7) << 3;
            *(v8h*)&Bps[buf][r * LDP + c] = *(const v8h*)(pT + (size_t)r * INDIM + k0 + c);
            *(v8h*)&Bns[buf][r * LDP + c] = *(const v8h*)(nT + (size_t)r * INDIM + k0 + c);
        }
#endif
    };

    auto compute = [&](int buf) {
#pragma unroll
        for (int s = 0; s < 2; ++s) {
            frag16 a[2], bp[2], bn[2];
#pragma unroll
            for (int mt = 0; mt < 2; ++mt) {
                int base = (wmi * 32 + mt * 16 + lrow) * LDP + s * 32;
                // A layout (matches B): lane-half h holds K = {h*16..h*16+15}
                a[mt].h[0] = *(const v8h*)(&As[buf][base + half * 16]);
                a[mt].h[1] = *(const v8h*)(&As[buf][base + half * 16 + 8]);
            }
#pragma unroll
            for (int nt = 0; nt < 2; ++nt) {
                int base = (wni * 32 + nt * 16 + lrow) * LDP + s * 32 + half * 16;
                bp[nt].h[0] = *(const v8h*)(&Bps[buf][base]);
                bp[nt].h[1] = *(const v8h*)(&Bps[buf][base + 8]);
                bn[nt].h[0] = *(const v8h*)(&Bns[buf][base]);
                bn[nt].h[1] = *(const v8h*)(&Bns[buf][base + 8]);
            }
#pragma unroll
            for (int mt = 0; mt < 2; ++mt)
#pragma unroll
                for (int nt = 0; nt < 2; ++nt) {
                    accP[mt][nt] = __builtin_amdgcn_wmma_f32_16x16x32_f16(
                        false, a[mt].v, false, bp[nt].v, (short)0, accP[mt][nt], false, false);
                    accN[mt][nt] = __builtin_amdgcn_wmma_f32_16x16x32_f16(
                        false, a[mt].v, false, bn[nt].v, (short)0, accN[mt][nt], false, false);
                }
        }
    };

    stage(0, 0);
    for (int ki = 0; ki < NKSTEP; ++ki) {
        int cur = ki & 1;
        if (ki + 1 < NKSTEP) stage(cur ^ 1, (ki + 1) * BK);   // overlap DMA with compute
#if USE_TDM
        if (wv == 0) {
            if (ki + 1 < NKSTEP) TWAIT(3);   // current tile's 3 loads done; next 3 in flight
            else                 TWAIT(0);
        }
#endif
        __syncthreads();
        compute(cur);
        __syncthreads();
    }

    // epilogue: D layout M = r + 8*half, N = lane%16
#pragma unroll
    for (int mt = 0; mt < 2; ++mt)
#pragma unroll
        for (int nt = 0; nt < 2; ++nt) {
            int col = col0 + wni * 32 + nt * 16 + lrow;
#pragma unroll
            for (int r = 0; r < 8; ++r) {
                int rowm = row0 + wmi * 32 + mt * 16 + half * 8 + r;
                float gp = g_pmos(accP[mt][nt][r]);
                float gn = g_pmos(accN[mt][nt][r]);
                out[(size_t)rowm * OUTDIM + col] = gp - gn;
            }
        }
}

// ---------------------------------------------------------------------------
// Fallback: fused conversion + GEMM (no workspace needed)
// ---------------------------------------------------------------------------
__global__ __launch_bounds__(256) void nmos_wmma_gemm_fused(const float* __restrict__ x,
                                                            const float* __restrict__ w,
                                                            float* __restrict__ out) {
    __shared__ __align__(16) _Float16 As[BM * LDP];
    __shared__ __align__(16) _Float16 Bp[BN * LDP];
    __shared__ __align__(16) _Float16 Bn[BN * LDP];

    const int t    = threadIdx.x;
    const int lane = t & 31;
    const int wv   = t >> 5;
    const int wmi  = wv >> 1;
    const int wni  = wv & 1;
    const int half = lane >> 4;
    const int lrow = lane & 15;

    const int row0 = blockIdx.y * BM;
    const int col0 = blockIdx.x * BN;

    v8f accP[2][2], accN[2][2];
#pragma unroll
    for (int mt = 0; mt < 2; ++mt)
#pragma unroll
        for (int nt = 0; nt < 2; ++nt)
#pragma unroll
            for (int r = 0; r < 8; ++r) {
                accP[mt][nt][r] = 0.0f;
                accN[mt][nt][r] = 0.0f;
            }

    for (int k0 = 0; k0 < INDIM; k0 += BK) {
#pragma unroll
        for (int it = 0; it < 8; ++it) {
            int v = it * 256 + t, r = v >> 4, c = (v & 15) << 2;
            float4 xv = *(const float4*)(x + (size_t)(row0 + r) * INDIM + k0 + c);
            v4h h;
            h[0] = (_Float16)f_nmos(xv.x);
            h[1] = (_Float16)f_nmos(xv.y);
            h[2] = (_Float16)f_nmos(xv.z);
            h[3] = (_Float16)f_nmos(xv.w);
            *(v4h*)(&As[r * LDP + c]) = h;
        }
#pragma unroll
        for (int it = 0; it < 4; ++it) {
            int v = it * 256 + t, r = v >> 4, c = (v & 15) << 2;
            float4 wv4 = *(const float4*)(w + (size_t)(col0 + r) * INDIM + k0 + c);
            v4h hp, hn;
            float q;
            q = fminf(fmaxf(rintf(wv4.x), -8.0f), 7.0f);
            hp[0] = (_Float16)fmaxf(q, 0.0f); hn[0] = (_Float16)fmaxf(-q, 0.0f);
            q = fminf(fmaxf(rintf(wv4.y), -8.0f), 7.0f);
            hp[1] = (_Float16)fmaxf(q, 0.0f); hn[1] = (_Float16)fmaxf(-q, 0.0f);
            q = fminf(fmaxf(rintf(wv4.z), -8.0f), 7.0f);
            hp[2] = (_Float16)fmaxf(q, 0.0f); hn[2] = (_Float16)fmaxf(-q, 0.0f);
            q = fminf(fmaxf(rintf(wv4.w), -8.0f), 7.0f);
            hp[3] = (_Float16)fmaxf(q, 0.0f); hn[3] = (_Float16)fmaxf(-q, 0.0f);
            *(v4h*)(&Bp[r * LDP + c]) = hp;
            *(v4h*)(&Bn[r * LDP + c]) = hn;
        }
        __syncthreads();
#pragma unroll
        for (int s = 0; s < 2; ++s) {
            frag16 a[2], bp[2], bn[2];
#pragma unroll
            for (int mt = 0; mt < 2; ++mt) {
                int base = (wmi * 32 + mt * 16 + lrow) * LDP + s * 32;
                a[mt].h[0] = *(const v8h*)(&As[base + half * 16]);
                a[mt].h[1] = *(const v8h*)(&As[base + half * 16 + 8]);
            }
#pragma unroll
            for (int nt = 0; nt < 2; ++nt) {
                int base = (wni * 32 + nt * 16 + lrow) * LDP + s * 32 + half * 16;
                bp[nt].h[0] = *(const v8h*)(&Bp[base]);
                bp[nt].h[1] = *(const v8h*)(&Bp[base + 8]);
                bn[nt].h[0] = *(const v8h*)(&Bn[base]);
                bn[nt].h[1] = *(const v8h*)(&Bn[base + 8]);
            }
#pragma unroll
            for (int mt = 0; mt < 2; ++mt)
#pragma unroll
                for (int nt = 0; nt < 2; ++nt) {
                    accP[mt][nt] = __builtin_amdgcn_wmma_f32_16x16x32_f16(
                        false, a[mt].v, false, bp[nt].v, (short)0, accP[mt][nt], false, false);
                    accN[mt][nt] = __builtin_amdgcn_wmma_f32_16x16x32_f16(
                        false, a[mt].v, false, bn[nt].v, (short)0, accN[mt][nt], false, false);
                }
        }
        __syncthreads();
    }

#pragma unroll
    for (int mt = 0; mt < 2; ++mt)
#pragma unroll
        for (int nt = 0; nt < 2; ++nt) {
            int col = col0 + wni * 32 + nt * 16 + lrow;
#pragma unroll
            for (int r = 0; r < 8; ++r) {
                int rowm = row0 + wmi * 32 + mt * 16 + half * 8 + r;
                float gp = g_pmos(accP[mt][nt][r]);
                float gn = g_pmos(accN[mt][nt][r]);
                out[(size_t)rowm * OUTDIM + col] = gp - gn;
            }
        }
}

extern "C" void kernel_launch(void* const* d_in, const int* in_sizes, int n_in,
                              void* d_out, int out_size, void* d_ws, size_t ws_size,
                              hipStream_t stream) {
    (void)in_sizes; (void)n_in; (void)out_size;
    const float* x = (const float*)d_in[0];   // (8192, 1024) f32
    const float* w = (const float*)d_in[1];   // (4096, 1024) f32
    float* out = (float*)d_out;               // (8192, 4096) f32

    const size_t need = (size_t)BDIM * INDIM * 2        // i_f16
                      + 2 * (size_t)OUTDIM * INDIM * 2; // wp_f16 + wn_f16
    dim3 grid(OUTDIM / BN, BDIM / BM);        // (64, 64)

    if (ws_size >= need) {
        _Float16* ih  = (_Float16*)d_ws;
        _Float16* wpp = ih + (size_t)BDIM * INDIM;
        _Float16* wnn = wpp + (size_t)OUTDIM * INDIM;
        prep_x_kernel<<<(BDIM * INDIM / 4) / 256, 256, 0, stream>>>(x, ih);
        prep_w_kernel<<<(OUTDIM * INDIM / 4) / 256, 256, 0, stream>>>(w, wpp, wnn);
        nmos_wmma_gemm_f16<<<grid, 256, 0, stream>>>(ih, wpp, wnn, out);
    } else {
        nmos_wmma_gemm_fused<<<grid, 256, 0, stream>>>(x, w, out);
    }
}
